// LSTMClassifier_807453851828
// MI455X (gfx1250) — matrix operand order repaired and checked by reading, not compiled, not run on hardware
//
#include <hip/hip_runtime.h>
#include <cstdint>
#include <cstddef>

// ---------------- types ----------------
typedef __bf16 bf16_t;
typedef __attribute__((ext_vector_type(16))) __bf16 v16bf;
typedef __attribute__((ext_vector_type(8)))  __bf16 v8bf;
typedef __attribute__((ext_vector_type(8)))  float  v8f;
typedef int v4i __attribute__((vector_size(16)));

#define AS1 __attribute__((address_space(1)))
#define AS3 __attribute__((address_space(3)))

#if defined(__gfx1250__) && __has_builtin(__builtin_amdgcn_global_load_async_to_lds_b128) && __has_builtin(__builtin_amdgcn_s_wait_asynccnt)
#define HAVE_ASYNC 1
#else
#define HAVE_ASYNC 0
#endif

// Problem dims
#define BB 64
#define SS 2048
#define VV 50257
#define EE 300
#define EP 320          // E padded to multiple of 32
#define HH 256
#define G4 1024         // 4*H

__device__ __forceinline__ float sigf(float x) { return 1.0f / (1.0f + __expf(-x)); }

__device__ __forceinline__ v8f wmma_bf16(v16bf a, v16bf b, v8f c) {
  return __builtin_amdgcn_wmma_f32_16x16x32_bf16(false, a, false, b, (short)0, c, false, false);
}

// A fragment (16x32 bf16) from a row-major bf16 buffer with given row stride.
// Lane l holds row m=l%16; K runs [ks, ks+16) with ks = 16*(l/16).
__device__ __forceinline__ v16bf load_a_frag(const bf16_t* base, int lane, int kbase, int rstride) {
  int m  = lane & 15;
  int ks = (lane >> 4) * 16;
  const bf16_t* p = base + m * rstride + kbase + ks;
  union { v16bf v; v8bf h[2]; } u;
  u.h[0] = *(const v8bf*)(p);
  u.h[1] = *(const v8bf*)(p + 8);
  return u.v;
}

// B fragment (32x16 bf16) from row-major (N,K) buffer: lane l holds column n=n0+l%16,
// 16 contiguous K starting at kbase + 16*(l/16).
__device__ __forceinline__ v16bf load_b_frag(const bf16_t* baseNK, int lane, int n0, int kbase, int kstride) {
  int n = n0 + (lane & 15);
  int k = kbase + ((lane >> 4) << 4);
  return *(const v16bf*)(baseNK + (long)n * kstride + k);
}

// ---------------- prep kernels: f32 -> bf16 (transposed/padded) ----------------
__global__ void k_cvt_emb(const float* __restrict__ emb, bf16_t* __restrict__ embB) {
  long i = (long)blockIdx.x * blockDim.x + threadIdx.x;
  if (i >= (long)VV * EP) return;
  long v = i / EP; int k = (int)(i % EP);
  embB[i] = (k < EE) ? (bf16_t)emb[v * EE + k] : (bf16_t)0.0f;
}
__global__ void k_cvt_wx(const float* __restrict__ Wx, bf16_t* __restrict__ WxT) {
  int i = blockIdx.x * blockDim.x + threadIdx.x;      // over 1024*320
  if (i >= G4 * EP) return;
  int n = i / EP, k = i % EP;
  WxT[i] = (k < EE) ? (bf16_t)Wx[(long)k * G4 + n] : (bf16_t)0.0f;
}
__global__ void k_cvt_wh(const float* __restrict__ Wh, bf16_t* __restrict__ WhT) {
  int i = blockIdx.x * blockDim.x + threadIdx.x;      // over 1024*256
  if (i >= G4 * HH) return;
  int n = i >> 8, k = i & 255;
  WhT[i] = (bf16_t)Wh[(long)k * G4 + n];
}

// ---------------- Phase 1: Zx[b*S+t][0:1024] = embed(gather) @ Wx  (bf16 WMMA) ---
// grid (1024, 8), block 512. Tile 128(M) x 128(N), K loop 10 chunks of 32.
__global__ __launch_bounds__(512) void k_phase1(
    const int*    __restrict__ inputs,
    const bf16_t* __restrict__ embB,   // [V][320]
    const bf16_t* __restrict__ WxT,    // [1024][320]  (N,K)
    bf16_t*       __restrict__ Zx)     // [B*S][1024]
{
  __shared__ bf16_t As[128 * 32];
  __shared__ bf16_t Bs[128 * 32];
  __shared__ int    idxs[128];

  const int tid  = threadIdx.x;
  const int lane = tid & 31;
  const int w    = tid >> 5;            // 0..15
  const long rowBase = (long)blockIdx.x * 128;
  const int  nBase   = blockIdx.y * 128;

  if (tid < 128) idxs[tid] = inputs[rowBase + tid];
  __syncthreads();

  const int mt    = w & 7;              // wave's M tile (0..7)
  const int nhalf = w >> 3;             // 0/1 -> N tiles [nhalf*4, nhalf*4+4)
  v8f acc[4] = {};

  for (int kc = 0; kc < 10; ++kc) {
    __syncthreads();
    // Stage: each of 512 threads moves one 16B chunk of A and one of B.
    {
      int r = tid >> 2, p = tid & 3;
      const bf16_t* sa = embB + (long)idxs[r] * EP + kc * 32 + p * 8;
      bf16_t*       da = As + r * 32 + p * 8;
      const bf16_t* sb = WxT + (long)(nBase + r) * EP + kc * 32 + p * 8;
      bf16_t*       db = Bs + r * 32 + p * 8;
#if HAVE_ASYNC
      __builtin_amdgcn_global_load_async_to_lds_b128((AS1 v4i*)sa, (AS3 v4i*)da, 0, 0);
      __builtin_amdgcn_global_load_async_to_lds_b128((AS1 v4i*)sb, (AS3 v4i*)db, 0, 0);
#else
      *(v8bf*)da = *(const v8bf*)sa;
      *(v8bf*)db = *(const v8bf*)sb;
#endif
    }
#if HAVE_ASYNC
    __builtin_amdgcn_s_wait_asynccnt(0);
#endif
    __syncthreads();

    v16bf a = load_a_frag(As + mt * 16 * 32, lane, 0, 32);
#pragma unroll
    for (int i = 0; i < 4; ++i) {
      int nt = nhalf * 4 + i;
      v16bf b = load_b_frag(Bs, lane, nt * 16, 0, 32);
      acc[i] = wmma_bf16(a, b, acc[i]);
    }
  }

  // Store D tiles: element (r,lane) -> m = r + 8*(lane>=16), n = lane&15
#pragma unroll
  for (int i = 0; i < 4; ++i) {
    int nt = nhalf * 4 + i;
    long col = nBase + nt * 16 + (lane & 15);
#pragma unroll
    for (int r = 0; r < 8; ++r) {
      long row = rowBase + mt * 16 + r + 8 * (lane >> 4);
      Zx[row * G4 + col] = (bf16_t)acc[i][r];
    }
  }
}

// ---------------- Phase 2: persistent LSTM recurrence ------------------------
// grid 4, block 512 (16 wave32). Each WG owns 16 batch rows.
// Wave w owns hidden slice j in [16w,16w+16); its 4 accumulator tiles are the
// i/f/g/o gate tiles for that slice (n = g*256 + j).  Wh(bf16) split:
// K-chunks 0..3 register-resident, K-chunks 4..7 in LDS (256KB).
__global__ __launch_bounds__(512) void k_phase2(
    const bf16_t* __restrict__ WhT,     // [1024][256] (N,K)
    const bf16_t* __restrict__ Zx,      // [B*S][1024]
    const float*  __restrict__ bias,    // [1024]
    const int*    __restrict__ lengths, // [64]
    float*        __restrict__ lastH)   // [64][256]
{
  extern __shared__ char smem[];
  bf16_t* Whs   = (bf16_t*)smem;                                   // [1024][128] K=128..255
  bf16_t* Hs    = (bf16_t*)(smem + (size_t)G4 * 128 * 2);          // [16][256]
  float*  biasS = (float*)(smem + (size_t)G4 * 128 * 2 + 16 * HH * 2);
  int*    lenS  = (int*)((char*)biasS + G4 * 4);

  const int tid  = threadIdx.x;
  const int lane = tid & 31;
  const int w    = tid >> 5;            // 0..15
  const int r0   = blockIdx.x * 16;     // batch rows [r0, r0+16)

  for (int i = tid; i < G4; i += 512) biasS[i] = bias[i];
  if (tid < 16) lenS[tid] = lengths[r0 + tid];
  for (int i = tid; i < 16 * HH; i += 512) Hs[i] = (bf16_t)0.0f;   // h0 = 0
  // Copy Wh K-half [128..255] into LDS: 16384 chunks of 8 elements
  for (int i = tid; i < G4 * 128 / 8; i += 512) {
    int n  = i >> 4;
    int c8 = (i & 15) * 8;
    *(v8bf*)(Whs + n * 128 + c8) = *(const v8bf*)(WhT + (long)n * HH + 128 + c8);
  }

  // Register-resident B fragments for K-chunks 0..3 (gates x kc = 16 frags = 128 VGPRs)
  const int j0 = w * 16;
  v16bf bfr[4][4];
#pragma unroll
  for (int g = 0; g < 4; ++g)
#pragma unroll
    for (int kc = 0; kc < 4; ++kc)
      bfr[g][kc] = load_b_frag(WhT, lane, g * HH + j0, kc * 32, HH);

  float c[8];
#pragma unroll
  for (int r = 0; r < 8; ++r) c[r] = 0.0f;

  const int jj = j0 + (lane & 15);      // this lane's hidden column
  const int mB = 8 * (lane >> 4);       // row offset of this lane's elements

  for (int t = 0; t < SS; ++t) {
    __syncthreads();                    // Hs from previous step ready
    v8f acc[4] = {};
#pragma unroll
    for (int kc = 0; kc < 8; ++kc) {
      v16bf a = load_a_frag(Hs, lane, kc * 32, HH);
#pragma unroll
      for (int g = 0; g < 4; ++g) {
        v16bf b;
        if (kc < 4) b = bfr[g][kc];
        else {
          int n  = g * HH + j0 + (lane & 15);
          int kk = (kc - 4) * 32 + ((lane >> 4) << 4);
          b = *(const v16bf*)(Whs + n * 128 + kk);
        }
        acc[g] = wmma_bf16(a, b, acc[g]);
      }
    }
    __syncthreads();                    // all WMMA reads of Hs done

    // Gate math: lane element r -> (m = r + mB, j = jj); acc[0..3] = i,f,g,o
#pragma unroll
    for (int r = 0; r < 8; ++r) {
      int m = r + mB;
      const bf16_t* zrow = Zx + ((long)(r0 + m) * SS + t) * G4;
      __builtin_prefetch(zrow + G4, 0, 0);     // next timestep's row
      float zi = acc[0][r] + (float)zrow[jj]            + biasS[jj];
      float zf = acc[1][r] + (float)zrow[HH + jj]       + biasS[HH + jj];
      float zg = acc[2][r] + (float)zrow[2 * HH + jj]   + biasS[2 * HH + jj];
      float zo = acc[3][r] + (float)zrow[3 * HH + jj]   + biasS[3 * HH + jj];
      float cc = sigf(zf) * c[r] + sigf(zi) * tanhf(zg);
      c[r] = cc;
      float h = sigf(zo) * tanhf(cc);
      Hs[m * HH + jj] = (bf16_t)h;
      if (t == lenS[m] - 1) lastH[(r0 + m) * HH + jj] = h;
    }
  }
}

// ---------------- Phase 3: head: out[b] = tanh(lastH@W1 + b1) @ W2 ------------
__global__ __launch_bounds__(256) void k_head(
    const float* __restrict__ lastH, const float* __restrict__ W1,
    const float* __restrict__ b1,    const float* __restrict__ W2,
    float* __restrict__ out)
{
  __shared__ float row[HH];
  __shared__ float red[HH];
  int b = blockIdx.x, j = threadIdx.x;
  row[j] = lastH[b * HH + j];
  __syncthreads();
  float acc = b1[j];
#pragma unroll 8
  for (int k = 0; k < HH; ++k) acc += row[k] * W1[k * HH + j];
  red[j] = tanhf(acc) * W2[j];
  __syncthreads();
  for (int s = 128; s > 0; s >>= 1) {
    if (j < s) red[j] += red[j + s];
    __syncthreads();
  }
  if (j == 0) out[b] = red[0];
}

// ---------------- host launch -------------------------------------------------
extern "C" void kernel_launch(void* const* d_in, const int* in_sizes, int n_in,
                              void* d_out, int out_size, void* d_ws, size_t ws_size,
                              hipStream_t stream) {
  (void)in_sizes; (void)n_in; (void)out_size; (void)ws_size;
  const int*   inputs  = (const int*)d_in[0];
  const int*   lengths = (const int*)d_in[1];
  const float* emb     = (const float*)d_in[2];
  const float* Wx      = (const float*)d_in[3];
  const float* Wh      = (const float*)d_in[4];
  const float* bvec    = (const float*)d_in[5];
  const float* W1      = (const float*)d_in[6];
  const float* b1      = (const float*)d_in[7];
  const float* W2      = (const float*)d_in[8];
  float* out = (float*)d_out;

  char* ws = (char*)d_ws;
  size_t off = 0;
  auto take = [&](size_t bytes) -> char* {
    char* p = ws + off;
    off += (bytes + 255) & ~(size_t)255;
    return p;
  };
  bf16_t* embB  = (bf16_t*)take((size_t)VV * EP * 2);       // 32.2 MB
  bf16_t* WxT   = (bf16_t*)take((size_t)G4 * EP * 2);       // 640 KB
  bf16_t* WhT   = (bf16_t*)take((size_t)G4 * HH * 2);       // 512 KB
  bf16_t* Zx    = (bf16_t*)take((size_t)BB * SS * G4 * 2);  // 256 MB
  float*  lastH = (float*)take((size_t)BB * HH * 4);        // 64 KB

  // weight/embedding conversion
  long ne = (long)VV * EP;
  k_cvt_emb<<<(unsigned)((ne + 255) / 256), 256, 0, stream>>>(emb, embB);
  k_cvt_wx<<<(G4 * EP + 255) / 256, 256, 0, stream>>>(Wx, WxT);
  k_cvt_wh<<<(G4 * HH + 255) / 256, 256, 0, stream>>>(Wh, WhT);

  // phase 1: big time-parallel GEMM
  k_phase1<<<dim3(BB * SS / 128, G4 / 128), 512, 0, stream>>>(inputs, embB, WxT, Zx);

  // phase 2: recurrence (4 WGPs, one per 16 batch rows)
  size_t smem2 = (size_t)G4 * 128 * 2 + 16 * HH * 2 + G4 * 4 + 16 * 4; // 274496 B
  (void)hipFuncSetAttribute((const void*)k_phase2, hipFuncAttributeMaxDynamicSharedMemorySize, (int)smem2);
  k_phase2<<<BB / 16, 512, smem2, stream>>>(WhT, Zx, bvec, lengths, lastH);

  // phase 3: head
  k_head<<<BB, 256, 0, stream>>>(lastH, W1, b1, W2, out);
}
